// Model_39676907882984
// MI455X (gfx1250) — matrix operand order repaired and checked by reading, not compiled, not run on hardware
//
#include <hip/hip_runtime.h>
#include <stdint.h>

typedef __attribute__((ext_vector_type(16))) __bf16 v16bf;
typedef __attribute__((ext_vector_type(8)))  float  v8f;
typedef __attribute__((ext_vector_type(4)))  unsigned int v4u;
typedef __attribute__((ext_vector_type(8)))  int    v8i;
typedef __attribute__((ext_vector_type(4)))  int    v4i;

#define DIM_S 2048
#define DIM_T 2048
#define DIM_D 64
#define INV_SCALE 10.0f                  /* 1 / 0.1 */
#define INV_KEEP (1.0f / 0.9f)
#define KEEP_THRESH 3865470566u          /* 0.9 * 2^32 */

struct U256 { uint4 a, b; };

// Assemble a 16x-bf16 WMMA operand from two 16-byte LDS chunks.
static __device__ __forceinline__ v16bf lds_ab(const unsigned short* sh, int i0, int i1) {
  U256 u;
  u.a = *(const uint4*)(sh + i0);
  u.b = *(const uint4*)(sh + i1);
  return __builtin_bit_cast(v16bf, u);
}

// float -> bf16 bits, round-to-nearest-even.
static __device__ __forceinline__ unsigned short f2bf(float f) {
  unsigned int u = __float_as_uint(f);
  u += 0x7FFFu + ((u >> 16) & 1u);
  return (unsigned short)(u >> 16);
}

static __device__ __forceinline__ unsigned int hash_u32(unsigned int x) {
  x ^= 0x2A2A2A2Au;             // "key 42" flavor
  x ^= x >> 16; x *= 0x7feb352du;
  x ^= x >> 15; x *= 0x846ca68bu;
  x ^= x >> 16;
  return x;
}

static __device__ __forceinline__ v8f wmma_bf16(v16bf a, v16bf b, v8f c) {
  return __builtin_amdgcn_wmma_f32_16x16x32_bf16(false, a, false, b, (short)0, c, false, false);
}

// Issue a TDM 2-D tile load: 32 rows x 64 fp32 from a 2048x64 fp32 tensor -> LDS.
// Descriptor per CDNA5 ISA (D# groups 0/1; groups 2/3 unused for 2-D).
// Toolchain builtin (clang-23 / therock-10.0): 6 args (v4u, v8i, v4i, v4i, v8i, cpol).
static __device__ __forceinline__ void tdm_load_tile(unsigned lds_addr,
                                                     unsigned long long gaddr) {
  v4u g0;
  g0[0] = 1u;                                            // count=1 (valid), user mode
  g0[1] = lds_addr;                                      // LDS byte address
  g0[2] = (unsigned)(gaddr & 0xFFFFFFFFull);             // global_addr[31:0]
  g0[3] = (unsigned)((gaddr >> 32) & 0x01FFFFFFull)      // global_addr[56:32]
        | (2u << 30);                                    // type = 2 ("image")
  v8i g1;
  g1[0] = (int)(2u << 16);                               // data_size = 2 (4 bytes)
  g1[1] = (int)(64u << 16);                              // tensor_dim0 = 64 (low16)
  g1[2] = (int)(2048u << 16);                            // tensor_dim1 = 2048 (low16)
  g1[3] = (int)(64u << 16);                              // tile_dim0 = 64
  g1[4] = 32;                                            // tile_dim1 = 32, tile_dim2 = 0
  g1[5] = 64;                                            // tensor_dim0_stride = 64
  g1[6] = 0;                                             // dim0_stride hi / dim1_stride lo
  g1[7] = 0;
  v4i gz4 = {0, 0, 0, 0};                                // groups 2/3: unused (2-D tile)
  v8i gz8 = {0, 0, 0, 0, 0, 0, 0, 0};
  __builtin_amdgcn_tensor_load_to_lds(g0, g1, gz4, gz4, gz8, 0);
}

__global__ __launch_bounds__(256)
void fa_fwd_kernel(const float* __restrict__ Qg,
                   const float* __restrict__ KVg,
                   float* __restrict__ Og) {
  // LDS: raw fp32 K/V tile (TDM double buffer), bf16 K row-major, bf16 V transposed,
  //      Q staging (reused per-wave as P scratch).
  __shared__ __align__(16) float          shKF[2 * 32 * 64];  // 16 KB (TDM ping-pong)
  __shared__ __align__(16) unsigned short shK[32 * 64];       // 4 KB
  __shared__ __align__(16) unsigned short shV[64 * 32];       // 4 KB
  __shared__ __align__(16) unsigned short shQ[128 * 64];      // 16 KB

  const int tid  = threadIdx.x;
  const int lane = tid & 31;
  const int w    = tid >> 5;          // wave id 0..7
  const int hi   = (lane >> 4) & 1;   // lane half
  const int ln   = lane & 15;

  const int bh      = blockIdx.x >> 4;       // fused batch*head 0..31
  const int rb      = blockIdx.x & 15;       // query row block 0..15
  const int qBase   = rb * 128;              // block's first query row
  const int rowBase = qBase + w * 16;        // wave's first query row

  const float* Qp = Qg  + (size_t)bh * DIM_S * DIM_D;
  const float* Kp = KVg + (size_t)bh * DIM_T * DIM_D;
  float*       Op = Og  + (size_t)bh * DIM_S * DIM_D;

  // LDS byte offsets of the two TDM buffers (generic-pointer low 32 bits = LDS offset).
  const unsigned kfOff0 = (unsigned)(uintptr_t)(void*)&shKF[0];
  const unsigned kfOff1 = (unsigned)(uintptr_t)(void*)&shKF[32 * 64];

  // ---- preload: wave 0 kicks the TDM for tile 0 while the block stages Q ----
  if (w == 0) tdm_load_tile(kfOff0, (unsigned long long)(uintptr_t)Kp);

  // ---------------- stage the 128x64 Q block to LDS as bf16 ----------------
  #pragma unroll
  for (int j = 0; j < 8; ++j) {
    int idx4 = tid + j * 256;
    int r = idx4 >> 4;
    int c = (idx4 & 15) * 4;
    float4 f = *(const float4*)(Qp + (size_t)(qBase + r) * DIM_D + c);
    unsigned s0 = f2bf(f.x), s1 = f2bf(f.y), s2 = f2bf(f.z), s3 = f2bf(f.w);
    *(uint2*)(shQ + r * 64 + c) = make_uint2(s0 | (s1 << 16), s2 | (s3 << 16));
  }
  __syncthreads();

  // Q as two WMMA A-operands (d = 0..31 and d = 32..63).
  // A 16x32 layout: lane row = ln; lane<16 holds K 0..7 & 16..23, lane>=16 holds 8..15 & 24..31.
  const int wq = w * 1024;
  v16bf Qa0 = lds_ab(shQ, wq + ln * 64 +  0 + hi * 8, wq + ln * 64 + 16 + hi * 8);
  v16bf Qa1 = lds_ab(shQ, wq + ln * 64 + 32 + hi * 8, wq + ln * 64 + 48 + hi * 8);

  unsigned short* shP = shQ + w * 1024;   // wave-private 16x32 bf16 P scratch

  v8f Oacc[4] = {};
  float mstate[8], lstate[8];
  #pragma unroll
  for (int v = 0; v < 8; ++v) { mstate[v] = -3.0e38f; lstate[v] = 0.0f; }

  #pragma unroll 1
  for (int it = 0; it < DIM_T / 32; ++it) {
    const int t0  = it * 32;
    const int buf = it & 1;

    // ---- wave 0: issue TDM for next tile, then guarantee current tile landed ----
    if (w == 0) {
      if (it + 1 < DIM_T / 32) {
        tdm_load_tile(buf ? kfOff0 : kfOff1,
                      (unsigned long long)(uintptr_t)(Kp + (size_t)(t0 + 32) * DIM_D));
        __builtin_amdgcn_s_wait_tensorcnt(1);   // in-order: tile `it` complete
      } else {
        __builtin_amdgcn_s_wait_tensorcnt(0);
      }
    }
    __syncthreads();   // publish tile `it`; previous iteration's shK/shV reads done

    // ---- convert fp32 tile (LDS, from TDM) -> bf16 shK (t x d) + shV (d x t) ----
    {
      int r = tid >> 3;            // t row 0..31
      int c = (tid & 7) * 8;       // d col 0..56
      const float* src = &shKF[buf * 2048 + r * 64 + c];
      float4 f0 = *(const float4*)(src);
      float4 f1 = *(const float4*)(src + 4);
      unsigned short b0 = f2bf(f0.x), b1 = f2bf(f0.y), b2 = f2bf(f0.z), b3 = f2bf(f0.w);
      unsigned short b4 = f2bf(f1.x), b5 = f2bf(f1.y), b6 = f2bf(f1.z), b7 = f2bf(f1.w);
      *(uint4*)(shK + r * 64 + c) =
          make_uint4((unsigned)b0 | ((unsigned)b1 << 16), (unsigned)b2 | ((unsigned)b3 << 16),
                     (unsigned)b4 | ((unsigned)b5 << 16), (unsigned)b6 | ((unsigned)b7 << 16));
      shV[(c + 0) * 32 + r] = b0; shV[(c + 1) * 32 + r] = b1;
      shV[(c + 2) * 32 + r] = b2; shV[(c + 3) * 32 + r] = b3;
      shV[(c + 4) * 32 + r] = b4; shV[(c + 5) * 32 + r] = b5;
      shV[(c + 6) * 32 + r] = b6; shV[(c + 7) * 32 + r] = b7;
    }
    __syncthreads();

    // ---- scores S[16x32] = Q(16x64) x K^T(64x32), two 16-col tiles ----
    // B 32x16 layout: lane col = ln; lane<16 holds K 0..7 & 16..23, lane>=16 holds 8..15 & 24..31.
    v8f Sc0 = {}, Sc1 = {};
    {
      v16bf Kb00 = lds_ab(shK, (0 * 16 + ln) * 64 +  0 + hi * 8, (0 * 16 + ln) * 64 + 16 + hi * 8);
      v16bf Kb01 = lds_ab(shK, (0 * 16 + ln) * 64 + 32 + hi * 8, (0 * 16 + ln) * 64 + 48 + hi * 8);
      v16bf Kb10 = lds_ab(shK, (1 * 16 + ln) * 64 +  0 + hi * 8, (1 * 16 + ln) * 64 + 16 + hi * 8);
      v16bf Kb11 = lds_ab(shK, (1 * 16 + ln) * 64 + 32 + hi * 8, (1 * 16 + ln) * 64 + 48 + hi * 8);
      Sc0 = wmma_bf16(Qa0, Kb00, Sc0);
      Sc0 = wmma_bf16(Qa1, Kb01, Sc0);
      Sc1 = wmma_bf16(Qa0, Kb10, Sc1);
      Sc1 = wmma_bf16(Qa1, Kb11, Sc1);
    }

    // ---- online softmax + dropout; write masked P to wave-private LDS ----
    #pragma unroll
    for (int v = 0; v < 8; ++v) {
      const int rv = v + 8 * hi;                    // C-layout row for this VGPR
      float s0 = Sc0[v] * INV_SCALE;
      float s1 = Sc1[v] * INV_SCALE;
      float mx = fmaxf(s0, s1);
      mx = fmaxf(mx, __shfl_xor(mx, 1));
      mx = fmaxf(mx, __shfl_xor(mx, 2));
      mx = fmaxf(mx, __shfl_xor(mx, 4));
      mx = fmaxf(mx, __shfl_xor(mx, 8));
      float mnew = fmaxf(mstate[v], mx);
      float corr = __expf(mstate[v] - mnew);
      float p0 = __expf(s0 - mnew);
      float p1 = __expf(s1 - mnew);
      float rs = p0 + p1;
      rs += __shfl_xor(rs, 1);
      rs += __shfl_xor(rs, 2);
      rs += __shfl_xor(rs, 4);
      rs += __shfl_xor(rs, 8);
      lstate[v] = lstate[v] * corr + rs;            // normalizer: pre-dropout probs
      mstate[v] = mnew;
      #pragma unroll
      for (int nt = 0; nt < 4; ++nt) Oacc[nt][v] *= corr;

      unsigned idx0 = (unsigned)((bh * DIM_S + rowBase + rv) * DIM_T + t0 + ln);
      unsigned idx1 = idx0 + 16u;
      float pm0 = (hash_u32(idx0) < KEEP_THRESH) ? p0 * INV_KEEP : 0.0f;
      float pm1 = (hash_u32(idx1) < KEEP_THRESH) ? p1 * INV_KEEP : 0.0f;
      shP[rv * 32 + ln]      = f2bf(pm0);
      shP[rv * 32 + 16 + ln] = f2bf(pm1);
    }

    // ---- O += P(16x32) x V(32x64), four 16-col d tiles ----
    {
      v16bf Pa = lds_ab(shP, ln * 32 + hi * 8, ln * 32 + 16 + hi * 8);
      #pragma unroll
      for (int nt = 0; nt < 4; ++nt) {
        v16bf Vb = lds_ab(shV, (nt * 16 + ln) * 32 + hi * 8,
                               (nt * 16 + ln) * 32 + 16 + hi * 8);
        Oacc[nt] = wmma_bf16(Pa, Vb, Oacc[nt]);
      }
    }
  }

  // ---------------- epilogue: normalize and store ----------------
  float rl[8];
  #pragma unroll
  for (int v = 0; v < 8; ++v) rl[v] = 1.0f / lstate[v];

  #pragma unroll
  for (int nt = 0; nt < 4; ++nt) {
    #pragma unroll
    for (int v = 0; v < 8; ++v) {
      const int rv = v + 8 * hi;
      Op[(size_t)(rowBase + rv) * DIM_D + nt * 16 + ln] = Oacc[nt][v] * rl[v];
    }
  }
}

extern "C" void kernel_launch(void* const* d_in, const int* in_sizes, int n_in,
                              void* d_out, int out_size, void* d_ws, size_t ws_size,
                              hipStream_t stream) {
  (void)in_sizes; (void)n_in; (void)out_size; (void)d_ws; (void)ws_size;
  const float* Q  = (const float*)d_in[0];
  const float* KV = (const float*)d_in[1];
  float* O = (float*)d_out;
  // 32 (B*H) fused heads x 16 row-blocks of 128 query rows; 256 threads = 8 wave32.
  dim3 grid(32 * 16), block(256);
  fa_fwd_kernel<<<grid, block, 0, stream>>>(Q, KV, O);
}
